// LNN_36842229465181
// MI455X (gfx1250) — matrix operand order repaired and checked by reading, not compiled, not run on hardware
//
#include <hip/hip_runtime.h>
#include <math.h>

#define NDOF 7
#define MDIM 28
#define WHID 64
#define SPB  16                 // samples per block
#define JROWS (SPB*NDOF)        // 112 jacobian rows
#define NTHREADS 128

typedef _Float16 half_t;
typedef half_t v16h __attribute__((ext_vector_type(16)));
typedef float  v8f  __attribute__((ext_vector_type(8)));
typedef unsigned int uint32x4 __attribute__((ext_vector_type(4)));
typedef int int32x4 __attribute__((ext_vector_type(4)));
typedef int int32x8 __attribute__((ext_vector_type(8)));

// np.tril_indices(7,-1) order
__device__ constexpr int TRI_I[21] = {1,2,2,3,3,3,4,4,4,4,5,5,5,5,5,6,6,6,6,6,6};
__device__ constexpr int TRI_J[21] = {0,0,1,0,1,2,0,1,2,3,0,1,2,3,4,0,1,2,3,4,5};

struct SharedMem {
    float  sW0[WHID*NDOF];        // current layer0 weight (WM0 then WV0), f32
    float  sB0[WHID];
    float  sB1[WHID];
    float  sB2[MDIM];
    float  sWv2[WHID];            // WV2 row
    float  sq[SPB*NDOF], sdq[SPB*NDOF], stau[SPB*NDOF];
    float  sY[SPB*MDIM];          // MLP-M output (feeds spd_exp)
    float  sDY[SPB*MDIM*NDOF];    // dy[s][m][k]; ALSO TDM stage for W2 (1792 f32)
    float  sU[SPB*NDOF*NDOF];     // u[s][k][i] = (dM_k @ dq)_i
    float  sM[SPB*NDOF*NDOF];     // exp(U)
    float  sG[SPB*NDOF];          // gravity vector
    float  sRHS[SPB*NDOF];        // tau - c - g
    __attribute__((aligned(32))) half_t hW1t[WHID*WHID];   // W1^T, B-fragment-swizzled, f16
    __attribute__((aligned(32))) half_t hW2t[WHID*32];     // W2^T, B-fragment-swizzled (n padded to 32)
    __attribute__((aligned(32))) half_t hH0s[SPB*WHID];    // h0, A-fragment-swizzled
    __attribute__((aligned(32))) half_t hH1s[SPB*WHID];    // h1, A-fragment-swizzled
    half_t hDG0[SPB*WHID];        // sigmoid(a0), linear [s][c]
    half_t hDG1[SPB*WHID];        // sigmoid(a1), linear [s][n]
    union {
        __attribute__((aligned(32))) half_t hDY1s[8*1024]; // dY1 tiles, swizzled
        float stageW[WHID*WHID];                           // TDM stage for W1/WV1
    } u;
};

__device__ inline float softplus_f(float a) {
    return (a > 20.f) ? a : log1pf(expf(a));
}
__device__ inline float sigmoid_clip(float a) {
    a = fminf(20.f, fmaxf(-20.f, a));
    return 1.f / (1.f + expf(-a));
}

// A-fragment swizzle (ISA 7.12.2, 16-bit A 16x32): element slot inside the
// per-lane v16h for matrix element (row m, k index kk in [0,64)).
// lane = m + 16*hi, buffer index = (kc*32 + lane)*16 + slot.
__device__ inline int swzA(int m, int kk) {
    int kc = kk >> 5, kw = kk & 31;
    int hi = (kw >> 3) & 1;
    int slot = (kw & 7) | ((kw & 16) >> 1);
    return (kc*32 + m + 16*hi)*16 + slot;
}

__device__ inline v8f wmma_zero() {
    v8f z = {0.f,0.f,0.f,0.f,0.f,0.f,0.f,0.f};
    return z;
}
__device__ inline v8f wmma16(v16h a, v16h b, v8f c) {
    return __builtin_amdgcn_wmma_f32_16x16x32_f16(false, a, false, b, (short)0, c, false, false);
}

// ---- Tensor Data Mover: DMA a [d1 x d0] f32 tile (row stride d0) into LDS ----
__device__ inline unsigned int lds_off(const void* p) {
    return (unsigned int)(unsigned long long)p;   // low 32 bits of LDS aperture addr
}
__device__ inline void tdm_load_2d(const void* gsrc, unsigned int ldsoff,
                                   unsigned int d0, unsigned int d1) {
    unsigned long long ga = (unsigned long long)gsrc;
    uint32x4 g0;
    g0[0] = 1u;                                                      // count=1
    g0[1] = ldsoff;                                                  // lds_addr
    g0[2] = (unsigned int)(ga & 0xffffffffull);                      // global[31:0]
    g0[3] = (unsigned int)((ga >> 32) & 0x1ffffffull) | (2u << 30);  // global[56:32]|type=2
    int32x8 g1;
    g1[0] = (int)(2u << 16);                              // data_size = 4B
    g1[1] = (int)((d0 & 0xffffu) << 16);                  // tensor_dim0 lo
    g1[2] = (int)((d0 >> 16) | ((d1 & 0xffffu) << 16));   // dim0 hi | dim1 lo
    g1[3] = (int)((d1 >> 16) | ((d0 & 0xffffu) << 16));   // dim1 hi | tile_dim0 = d0
    g1[4] = (int)(d1 & 0xffffu);                          // tile_dim1 = d1 (tile_dim2=0)
    g1[5] = (int)d0;                                      // tensor_dim0_stride lo
    g1[6] = 0;
    g1[7] = 0;
    int32x4 z4 = {0,0,0,0};
#if __clang_major__ >= 23
    int32x8 z8 = {0,0,0,0,0,0,0,0};
    __builtin_amdgcn_tensor_load_to_lds(g0, g1, z4, z4, z8, 0);
#else
    __builtin_amdgcn_tensor_load_to_lds(g0, g1, z4, z4, 0);
#endif
}

// ---- forward 64x64 GEMM: a1 = h0 @ W1^T (+b1) -> softplus/sigmoid ----------
__device__ inline void fwd_gemm64(SharedMem& sm, int lane, int wave, bool storeH) {
    const int n0 = wave * 16;
    v16h a0 = *(const v16h*)&sm.hH0s[lane*16];
    v16h a1 = *(const v16h*)&sm.hH0s[(32 + lane)*16];
    v16h b0 = *(const v16h*)&sm.hW1t[lane*WHID + n0];
    v16h b1 = *(const v16h*)&sm.hW1t[(32 + lane)*WHID + n0];
    v8f acc = wmma16(a1, b1, wmma16(a0, b0, wmma_zero()));
    const int n = n0 + (lane & 15), hi = lane >> 4;
#pragma unroll
    for (int r = 0; r < 8; ++r) {
        int s = r + 8*hi;
        float v = acc[r] + sm.sB1[n];
        if (storeH) sm.hH1s[swzA(s, n)] = (half_t)softplus_f(v);
        sm.hDG1[s*WHID + n] = (half_t)sigmoid_clip(v);
    }
}

// ---- Jacobian GEMM-1: dY1 = dg1 .* (W1 @ (dg0 .* W0)), rows (s,k) ----------
__device__ inline void jacobian_gemm1(SharedMem& sm, int lane, int wave) {
    v16h bf[8];                       // all W1 B-fragments, reused across row tiles
#pragma unroll
    for (int kc = 0; kc < 2; ++kc)
#pragma unroll
        for (int ct = 0; ct < 4; ++ct)
            bf[kc*4 + ct] = *(const v16h*)&sm.hW1t[(kc*32 + lane)*WHID + ct*16];
    const int m = lane & 15, hi = lane >> 4;
    for (int rt = wave; rt < 8; rt += 4) {
        const int row = rt*16 + m;
        v16h a0, a1;
        if (row < JROWS) {
            const int s = row / NDOF, kd = row % NDOF;
#pragma unroll
            for (int e = 0; e < 16; ++e) {
                int j = e >> 1;
                int kb = ((j < 4) ? (2*j) : (16 + 2*(j - 4))) + 8*hi + (e & 1);
                a0[e] = (half_t)((float)sm.hDG0[s*WHID + kb]      * sm.sW0[kb*NDOF + kd]);
                a1[e] = (half_t)((float)sm.hDG0[s*WHID + kb + 32] * sm.sW0[(kb + 32)*NDOF + kd]);
            }
        } else {
#pragma unroll
            for (int e = 0; e < 16; ++e) { a0[e] = (half_t)0.f; a1[e] = (half_t)0.f; }
        }
#pragma unroll
        for (int ct = 0; ct < 4; ++ct) {
            v8f acc = wmma16(a1, bf[4 + ct], wmma16(a0, bf[ct], wmma_zero()));
            const int n = ct*16 + (lane & 15);
#pragma unroll
            for (int r = 0; r < 8; ++r) {
                int m2 = r + 8*hi;
                int grow = rt*16 + m2;
                float v = 0.f;
                if (grow < JROWS) v = acc[r] * (float)sm.hDG1[(grow / NDOF)*WHID + n];
                sm.u.hDY1s[rt*1024 + swzA(m2, n)] = (half_t)v;   // pad rows -> 0
            }
        }
    }
}

__global__ __launch_bounds__(NTHREADS)
void lnn_fused_kernel(const float* __restrict__ q,   const float* __restrict__ dq,
                      const float* __restrict__ tau,
                      const float* __restrict__ WM0, const float* __restrict__ bM0,
                      const float* __restrict__ WM1, const float* __restrict__ bM1,
                      const float* __restrict__ WM2, const float* __restrict__ bM2,
                      const float* __restrict__ WV0, const float* __restrict__ bV0,
                      const float* __restrict__ WV1, const float* __restrict__ bV1,
                      const float* __restrict__ WV2,
                      float* __restrict__ out, int Btot)
{
    __shared__ SharedMem sm;
    const int tid  = threadIdx.x;
    const int lane = tid & 31;
    const int wave = tid >> 5;
    const int s0   = blockIdx.x * SPB;

    // ---------------- P0: TDM big weights; VALU-load small ones ---------------
    if (wave == 0) {
        tdm_load_2d(WM1, lds_off(&sm.u.stageW[0]), WHID, WHID);   // 64x64 f32 -> stage
        tdm_load_2d(WM2, lds_off(&sm.sDY[0]), MDIM*WHID, 1);      // 1792 f32 -> sDY stage
    }
    if (tid == 0) { __builtin_prefetch(WV1, 0, 0); }
    for (int i = tid; i < WHID*NDOF; i += NTHREADS) sm.sW0[i] = WM0[i];
    for (int i = tid; i < WHID; i += NTHREADS) {
        sm.sB0[i] = bM0[i]; sm.sB1[i] = bM1[i]; sm.sWv2[i] = WV2[i];
    }
    for (int i = tid; i < MDIM; i += NTHREADS) sm.sB2[i] = bM2[i];
    for (int i = tid; i < SPB*NDOF; i += NTHREADS) {
        int s = i / NDOF, d = i % NDOF;
        int sg = s0 + s; if (sg >= Btot) sg = Btot - 1;
        sm.sq[i]   = q[sg*NDOF + d];
        sm.sdq[i]  = dq[sg*NDOF + d];
        sm.stau[i] = tau[sg*NDOF + d];
    }
    if (wave == 0) __builtin_amdgcn_s_wait_tensorcnt(0);
    __syncthreads();
    // convert staged f32 weights to f16 transposed WMMA-B layouts
    for (int i = tid; i < WHID*WHID; i += NTHREADS) {
        int rw = i >> 6, cl = i & 63, l = rw & 31, h = l >> 4, e = cl & 15;
        int kk = (rw & 32) + ((e < 8) ? (8*h + e) : (16 + 8*h + e - 8)), n = (cl & ~15) + (l & 15);
        sm.hW1t[i] = (half_t)sm.u.stageW[n*WHID + kk];
    }
    for (int i = tid; i < WHID*32; i += NTHREADS) {
        int rw = i >> 5, cl = i & 31, l = rw & 31, h = l >> 4, e = cl & 15;
        int kk = (rw & 32) + ((e < 8) ? (8*h + e) : (16 + 8*h + e - 8)), n = (cl & ~15) + (l & 15);
        sm.hW2t[i] = (n < MDIM) ? (half_t)sm.sDY[n*WHID + kk] : (half_t)0.f;
    }
    __syncthreads();

    // ---------------- P1: layer-0 (7->64), VALU ------------------------------
    for (int i = tid; i < SPB*WHID; i += NTHREADS) {
        int s = i >> 6, o = i & 63;
        float a = sm.sB0[o];
#pragma unroll
        for (int d = 0; d < NDOF; ++d) a += sm.sW0[o*NDOF + d] * sm.sq[s*NDOF + d];
        sm.hH0s[swzA(s, o)] = (half_t)softplus_f(a);
        sm.hDG0[i] = (half_t)sigmoid_clip(a);
    }
    __syncthreads();

    // ---------------- P2: WMMA a1 = h0 @ W1^T --------------------------------
    fwd_gemm64(sm, lane, wave, /*storeH=*/true);
    __syncthreads();

    // ---------------- P3: WMMA y = h1 @ W2^T (waves 0..1) --------------------
    if (wave < 2) {
        const int n0 = wave * 16;
        v16h a0 = *(const v16h*)&sm.hH1s[lane*16];
        v16h a1 = *(const v16h*)&sm.hH1s[(32 + lane)*16];
        v16h b0 = *(const v16h*)&sm.hW2t[lane*32 + n0];
        v16h b1 = *(const v16h*)&sm.hW2t[(32 + lane)*32 + n0];
        v8f acc = wmma16(a1, b1, wmma16(a0, b0, wmma_zero()));
        const int n = n0 + (lane & 15), hi = lane >> 4;
        if (n < MDIM) {
#pragma unroll
            for (int r = 0; r < 8; ++r) sm.sY[(r + 8*hi)*MDIM + n] = acc[r] + sm.sB2[n];
        }
    }
    // ---------------- P4: WMMA Jacobian GEMM-1 (M network) -------------------
    jacobian_gemm1(sm, lane, wave);
    __syncthreads();

    // ---------------- P5: WMMA Jacobian GEMM-2: dy = W2 @ dY1 ----------------
    {
        v16h bf[4];
#pragma unroll
        for (int kc = 0; kc < 2; ++kc)
#pragma unroll
            for (int ct = 0; ct < 2; ++ct)
                bf[kc*2 + ct] = *(const v16h*)&sm.hW2t[(kc*32 + lane)*32 + ct*16];
        const int hi = lane >> 4;
        for (int rt = wave; rt < 8; rt += 4) {
            v16h a0 = *(const v16h*)&sm.u.hDY1s[rt*1024 + lane*16];
            v16h a1 = *(const v16h*)&sm.u.hDY1s[rt*1024 + (32 + lane)*16];
#pragma unroll
            for (int ct = 0; ct < 2; ++ct) {
                v8f acc = wmma16(a1, bf[2 + ct], wmma16(a0, bf[ct], wmma_zero()));
                const int n = ct*16 + (lane & 15);
#pragma unroll
                for (int r = 0; r < 8; ++r) {
                    int grow = rt*16 + r + 8*hi;
                    if (grow < JROWS && n < MDIM) {
                        int s = grow / NDOF, k = grow % NDOF;
                        sm.sDY[s*(MDIM*NDOF) + n*NDOF + k] = acc[r];
                    }
                }
            }
        }
    }
    __syncthreads();

    // ---------------- P6: V-network weights (TDM WV1, VALU small) ------------
    if (wave == 0) tdm_load_2d(WV1, lds_off(&sm.u.stageW[0]), WHID, WHID);
    for (int i = tid; i < WHID*NDOF; i += NTHREADS) sm.sW0[i] = WV0[i];
    for (int i = tid; i < WHID; i += NTHREADS) { sm.sB0[i] = bV0[i]; sm.sB1[i] = bV1[i]; }
    if (wave == 0) __builtin_amdgcn_s_wait_tensorcnt(0);
    __syncthreads();
    for (int i = tid; i < WHID*WHID; i += NTHREADS) {
        int rw = i >> 6, cl = i & 63, l = rw & 31, h = l >> 4, e = cl & 15;
        int kk = (rw & 32) + ((e < 8) ? (8*h + e) : (16 + 8*h + e - 8)), n = (cl & ~15) + (l & 15);
        sm.hW1t[i] = (half_t)sm.u.stageW[n*WHID + kk];
    }
    __syncthreads();

    // ---------------- P7: V layer-0 ------------------------------------------
    for (int i = tid; i < SPB*WHID; i += NTHREADS) {
        int s = i >> 6, o = i & 63;
        float a = sm.sB0[o];
#pragma unroll
        for (int d = 0; d < NDOF; ++d) a += sm.sW0[o*NDOF + d] * sm.sq[s*NDOF + d];
        sm.hH0s[swzA(s, o)] = (half_t)softplus_f(a);
        sm.hDG0[i] = (half_t)sigmoid_clip(a);
    }
    __syncthreads();

    // ---------------- P8: WMMA aV1 (only dg1 needed) -------------------------
    fwd_gemm64(sm, lane, wave, /*storeH=*/false);
    __syncthreads();

    // ---------------- P9: WMMA V-Jacobian GEMM-1 -> hDY1s --------------------
    jacobian_gemm1(sm, lane, wave);
    __syncthreads();

    // ---------------- P10: g[s][k] = WV2 . dVY1[(s,k)] -----------------------
    if (tid < JROWS) {
        const int rt = tid >> 4, m = tid & 15;
        float acc = 0.f;
        for (int c = 0; c < WHID; ++c)
            acc += sm.sWv2[c] * (float)sm.u.hDY1s[rt*1024 + swzA(m, c)];
        sm.sG[tid] = acc;
    }
    __syncthreads();

    // ---------------- P11: per-(s,k): eig(U), exp, Daleckii-Krein, u_k -------
    if (tid < JROWS) {
        const int s = tid / NDOF, k = tid % NDOF;
        float A[7][7], V[7][7];
#pragma unroll
        for (int i = 0; i < 7; ++i)
#pragma unroll
            for (int j = 0; j < 7; ++j) V[i][j] = (i == j) ? 1.f : 0.f;
#pragma unroll
        for (int d = 0; d < 7; ++d) A[d][d] = sm.sY[s*MDIM + d];
#pragma unroll
        for (int o = 0; o < 21; ++o) {
            float v = sm.sY[s*MDIM + 7 + o];
            A[TRI_I[o]][TRI_J[o]] = v; A[TRI_J[o]][TRI_I[o]] = v;
        }
        for (int sw = 0; sw < 8; ++sw) {          // cyclic Jacobi
#pragma unroll
            for (int p = 0; p < 6; ++p) {
#pragma unroll
                for (int qq = p + 1; qq < 7; ++qq) {
                    float apq = A[p][qq];
                    float t = 0.f;
                    if (fabsf(apq) > 1e-30f) {
                        float th = (A[qq][qq] - A[p][p]) / (2.f * apq);
                        t = (th >= 0.f ? 1.f : -1.f) / (fabsf(th) + sqrtf(1.f + th*th));
                    }
                    if (t != 0.f) {
                        float cc = 1.f / sqrtf(1.f + t*t);
                        float ss = t * cc;
#pragma unroll
                        for (int i = 0; i < 7; ++i) {
                            float aip = A[i][p], aiq = A[i][qq];
                            A[i][p] = cc*aip - ss*aiq; A[i][qq] = ss*aip + cc*aiq;
                        }
#pragma unroll
                        for (int i = 0; i < 7; ++i) {
                            float api = A[p][i], aqi = A[qq][i];
                            A[p][i] = cc*api - ss*aqi; A[qq][i] = ss*api + cc*aqi;
                        }
#pragma unroll
                        for (int i = 0; i < 7; ++i) {
                            float vip = V[i][p], viq = V[i][qq];
                            V[i][p] = cc*vip - ss*viq; V[i][qq] = ss*vip + cc*viq;
                        }
                    }
                }
            }
        }
        float e[7], ex[7];
#pragma unroll
        for (int i = 0; i < 7; ++i) { e[i] = A[i][i]; ex[i] = expf(e[i]); }
        if (k == 0) {                             // M = V diag(exp e) V^T
#pragma unroll
            for (int i = 0; i < 7; ++i)
#pragma unroll
                for (int j = 0; j < 7; ++j) {
                    float acc = 0.f;
#pragma unroll
                    for (int c = 0; c < 7; ++c) acc += ex[c] * V[i][c] * V[j][c];
                    sm.sM[s*49 + i*7 + j] = acc;
                }
        }
        float Ak[7][7];                           // A_k = dU/dq_k from dy
#pragma unroll
        for (int d = 0; d < 7; ++d) Ak[d][d] = sm.sDY[s*(MDIM*NDOF) + d*NDOF + k];
#pragma unroll
        for (int o = 0; o < 21; ++o) {
            float v = sm.sDY[s*(MDIM*NDOF) + (7 + o)*NDOF + k];
            Ak[TRI_I[o]][TRI_J[o]] = v; Ak[TRI_J[o]][TRI_I[o]] = v;
        }
        float Wt[7][7], S[7][7];                  // S = (V^T Ak V) .* R
#pragma unroll
        for (int a2 = 0; a2 < 7; ++a2)
#pragma unroll
            for (int j = 0; j < 7; ++j) {
                float acc = 0.f;
#pragma unroll
                for (int i = 0; i < 7; ++i) acc += V[i][a2] * Ak[i][j];
                Wt[a2][j] = acc;
            }
#pragma unroll
        for (int a2 = 0; a2 < 7; ++a2)
#pragma unroll
            for (int b2 = 0; b2 < 7; ++b2) {
                float acc = 0.f;
#pragma unroll
                for (int j = 0; j < 7; ++j) acc += Wt[a2][j] * V[j][b2];
                float r;
                if (a2 == b2) r = ex[a2];
                else { float bd = e[a2] - e[b2]; r = (bd != 0.f) ? (ex[a2] - ex[b2]) / bd : 0.f; }
                S[a2][b2] = acc * r;
            }
#pragma unroll
        for (int i = 0; i < 7; ++i)               // dMk = V S V^T ; u_k = dMk dq
#pragma unroll
            for (int b2 = 0; b2 < 7; ++b2) {
                float acc = 0.f;
#pragma unroll
                for (int a2 = 0; a2 < 7; ++a2) acc += V[i][a2] * S[a2][b2];
                Wt[i][b2] = acc;
            }
#pragma unroll
        for (int i = 0; i < 7; ++i) {
            float acc = 0.f;
#pragma unroll
            for (int j = 0; j < 7; ++j) {
                float dm = 0.f;
#pragma unroll
                for (int b2 = 0; b2 < 7; ++b2) dm += Wt[i][b2] * V[j][b2];
                acc += dm * sm.sdq[s*NDOF + j];
            }
            sm.sU[s*49 + k*7 + i] = acc;
        }
    }
    __syncthreads();

    // ---------------- P12: c_i = sum_k dq_k u_k[i] - 0.5 dq.u_i --------------
    if (tid < JROWS) {
        const int s = tid / NDOF, i = tid % NDOF;
        float c1 = 0.f, w = 0.f;
#pragma unroll
        for (int k = 0; k < 7; ++k) {
            c1 += sm.sdq[s*NDOF + k] * sm.sU[s*49 + k*7 + i];
            w  += sm.sU[s*49 + i*7 + k] * sm.sdq[s*NDOF + k];
        }
        sm.sRHS[tid] = sm.stau[tid] - (c1 - 0.5f * w) - sm.sG[tid];
    }
    __syncthreads();

    // ---------------- P13: Cholesky solve M ddq = rhs ------------------------
    if (tid < SPB) {
        const int s = tid;
        float L[7][7], x[7];
#pragma unroll
        for (int i = 0; i < 7; ++i)
#pragma unroll
            for (int j = 0; j < 7; ++j) L[i][j] = sm.sM[s*49 + i*7 + j];
#pragma unroll
        for (int j = 0; j < 7; ++j) {
            float d = L[j][j];
#pragma unroll
            for (int k2 = 0; k2 < 7; ++k2) if (k2 < j) d -= L[j][k2]*L[j][k2];
            d = sqrtf(fmaxf(d, 1e-20f));
            L[j][j] = d;
#pragma unroll
            for (int i = 0; i < 7; ++i) if (i > j) {
                float v = L[i][j];
#pragma unroll
                for (int k2 = 0; k2 < 7; ++k2) if (k2 < j) v -= L[i][k2]*L[j][k2];
                L[i][j] = v / d;
            }
        }
#pragma unroll
        for (int i = 0; i < 7; ++i) {
            float v = sm.sRHS[s*NDOF + i];
#pragma unroll
            for (int k2 = 0; k2 < 7; ++k2) if (k2 < i) v -= L[i][k2]*x[k2];
            x[i] = v / L[i][i];
        }
#pragma unroll
        for (int i = 6; i >= 0; --i) {
            float v = x[i];
#pragma unroll
            for (int k2 = 0; k2 < 7; ++k2) if (k2 > i) v -= L[k2][i]*x[k2];
            x[i] = v / L[i][i];
        }
        if (s0 + s < Btot) {
#pragma unroll
            for (int i = 0; i < 7; ++i) out[(s0 + s)*NDOF + i] = x[i];
        }
    }
}

extern "C" void kernel_launch(void* const* d_in, const int* in_sizes, int n_in,
                              void* d_out, int out_size, void* d_ws, size_t ws_size,
                              hipStream_t stream) {
    (void)n_in; (void)d_ws; (void)ws_size; (void)out_size;
    const float* q   = (const float*)d_in[0];
    const float* dq  = (const float*)d_in[1];
    const float* tau = (const float*)d_in[2];
    const float* WM0 = (const float*)d_in[3];
    const float* bM0 = (const float*)d_in[4];
    const float* WM1 = (const float*)d_in[5];
    const float* bM1 = (const float*)d_in[6];
    const float* WM2 = (const float*)d_in[7];
    const float* bM2 = (const float*)d_in[8];
    const float* WV0 = (const float*)d_in[9];
    const float* bV0 = (const float*)d_in[10];
    const float* WV1 = (const float*)d_in[11];
    const float* bV1 = (const float*)d_in[12];
    const float* WV2 = (const float*)d_in[13];
    // d_in[14] = bV2: additive constant on V, does not affect g -> unused.

    const int Btot = in_sizes[0] / NDOF;
    const int nblocks = (Btot + SPB - 1) / SPB;
    lnn_fused_kernel<<<nblocks, NTHREADS, 0, stream>>>(
        q, dq, tau, WM0, bM0, WM1, bM1, WM2, bM2, WV0, bV0, WV1, bV1, WV2,
        (float*)d_out, Btot);
}
